// MinCutPoolLayer_25847113187666
// MI455X (gfx1250) — matrix operand order repaired and checked by reading, not compiled, not run on hardware
//
#include <hip/hip_runtime.h>

#define BB 8
#define NN 4096
#define FF 512
#define KK 100
#define KP 112
#define EPSV 1e-15f
#define BNEPS 1e-5f

typedef __attribute__((ext_vector_type(16))) __bf16 v16bf;
typedef __attribute__((ext_vector_type(8)))  float v8f;

union Frag {
    v16bf v;
    unsigned short u[16];
    uint4 q[2];
};

static __device__ inline unsigned short f2bf(float f) {
    unsigned int x = __float_as_uint(f);
    x += 0x7fffu + ((x >> 16) & 1u);   // round-to-nearest-even
    return (unsigned short)(x >> 16);
}

static __device__ inline unsigned int pack2(float a, float b) {
    return (unsigned int)f2bf(a) | ((unsigned int)f2bf(b) << 16);
}

static __device__ inline v8f zero_v8f() {
    v8f z;
    for (int i = 0; i < 8; ++i) z[i] = 0.0f;
    return z;
}

static __device__ inline void cvt16(Frag& a, float4 f0, float4 f1, float4 f2, float4 f3) {
    a.u[0]  = f2bf(f0.x); a.u[1]  = f2bf(f0.y); a.u[2]  = f2bf(f0.z); a.u[3]  = f2bf(f0.w);
    a.u[4]  = f2bf(f1.x); a.u[5]  = f2bf(f1.y); a.u[6]  = f2bf(f1.z); a.u[7]  = f2bf(f1.w);
    a.u[8]  = f2bf(f2.x); a.u[9]  = f2bf(f2.y); a.u[10] = f2bf(f2.z); a.u[11] = f2bf(f2.w);
    a.u[12] = f2bf(f3.x); a.u[13] = f2bf(f3.y); a.u[14] = f2bf(f3.z); a.u[15] = f2bf(f3.w);
}

// ---------------------------------------------------------------------------
// Kernel 1: s = softmax(x @ W^T + b), one wave per 16 rows.
// A-frag direct from x (fp32->bf16 in regs); B-frag direct from W rows.
// Writes s bf16 K-major: s16t[b][k][n] (pad k>=100 zero), and rowsq = sum_k s^2.
// ---------------------------------------------------------------------------
__global__ __launch_bounds__(32) void k_s_softmax(const float* __restrict__ x,
                                                  const float* __restrict__ W,
                                                  const float* __restrict__ bias,
                                                  float* __restrict__ rowsq,
                                                  float* __restrict__ den,
                                                  unsigned short* __restrict__ s16t) {
    __shared__ float Slog[16][KP];
    int lane = threadIdx.x;
    int row0 = blockIdx.x * 16;                 // flattened b*NN + n
    int b = row0 / NN;
    int n0 = row0 % NN;
    int m = lane & 15, hi = lane >> 4;

    if (blockIdx.x < BB && lane == 0) den[blockIdx.x] = 0.f;  // init for k_as

    v8f acc[7];
    for (int c = 0; c < 7; ++c) acc[c] = zero_v8f();

    const float* xr = x + (size_t)(row0 + m) * FF;

    for (int kk = 0; kk < FF; kk += 32) {
        Frag a;
        const float* p0 = xr + kk + hi * 8;
        float4 f0 = *(const float4*)(p0);
        float4 f1 = *(const float4*)(p0 + 4);
        float4 f2 = *(const float4*)(p0 + 16);
        float4 f3 = *(const float4*)(p0 + 20);
        cvt16(a, f0, f1, f2, f3);
        for (int c = 0; c < 7; ++c) {
            int kcol = c * 16 + m;
            Frag bf;
            if (kcol < KK) {
                const float* wp = W + (size_t)kcol * FF + kk + hi * 8;
                float4 g0 = *(const float4*)(wp);
                float4 g1 = *(const float4*)(wp + 4);
                float4 g2 = *(const float4*)(wp + 16);
                float4 g3 = *(const float4*)(wp + 20);
                cvt16(bf, g0, g1, g2, g3);
            } else {
                for (int j = 0; j < 16; ++j) bf.u[j] = 0;
            }
            acc[c] = __builtin_amdgcn_wmma_f32_16x16x32_bf16(
                false, a.v, false, bf.v, (short)0, acc[c], false, false);
        }
    }

    for (int c = 0; c < 7; ++c) {               // dump logits + bias
        int kcol = c * 16 + m;
        float bb = (kcol < KK) ? bias[kcol] : 0.f;
        for (int r = 0; r < 8; ++r) Slog[r + hi * 8][kcol] = acc[c][r] + bb;
    }
    __syncthreads();

    if (lane < 16) {                             // softmax, one row per lane
        float mx = -1e30f;
        for (int k = 0; k < KK; ++k) mx = fmaxf(mx, Slog[lane][k]);
        float sum = 0.f;
        for (int k = 0; k < KK; ++k) {
            float e = __expf(Slog[lane][k] - mx);
            Slog[lane][k] = e;
            sum += e;
        }
        float inv = 1.f / sum;
        float sq = 0.f;
        for (int k = 0; k < KK; ++k) {
            float sv = Slog[lane][k] * inv;
            Slog[lane][k] = sv;
            sq += sv * sv;
        }
        rowsq[row0 + lane] = sq;
    }
    __syncthreads();

    // cooperative K-major bf16 write: s16t[b][k][n0..n0+15] (32B per k)
    unsigned short* Sb = s16t + (size_t)b * KP * NN;
    for (int k = lane; k < KP; k += 32) {
        unsigned int w[8];
        if (k < KK) {
            for (int i = 0; i < 8; ++i)
                w[i] = pack2(Slog[2 * i][k], Slog[2 * i + 1][k]);
        } else {
            for (int i = 0; i < 8; ++i) w[i] = 0u;
        }
        uint4 qa; qa.x = w[0]; qa.y = w[1]; qa.z = w[2]; qa.w = w[3];
        uint4 qb; qb.x = w[4]; qb.y = w[5]; qb.z = w[6]; qb.w = w[7];
        uint4* dp = (uint4*)(Sb + (size_t)k * NN + n0);
        dp[0] = qa; dp[1] = qb;
    }
}

// ---------------------------------------------------------------------------
// Kernel 2: As = adj @ s  (dominant GEMM; adj streamed from HBM exactly once).
// LDS-free: A-frag from adj rows (fp32->bf16 in regs), B-frags direct b128
// from L2-resident K-major s. Fused deg row-sums -> mincut_den accumulation.
// Output As bf16 K-major with one b128 store per tile per lane.
// ---------------------------------------------------------------------------
__global__ __launch_bounds__(256) void k_as(const float* __restrict__ adj,
                                            const unsigned short* __restrict__ s16t,
                                            const float* __restrict__ rowsq,
                                            float* __restrict__ den,
                                            unsigned short* __restrict__ as16t) {
    int lane = threadIdx.x & 31;
    int wv = threadIdx.x >> 5;
    int flat = (blockIdx.x * 8 + wv) * 16;       // flattened node row base
    int b = flat / NN;
    int n0 = flat % NN;
    int m = lane & 15, hi = lane >> 4;

    const float* Arow = adj + ((size_t)b * NN + n0 + m) * NN;
    const unsigned short* Sb = s16t + (size_t)b * KP * NN;

    v8f acc[7];
    for (int c = 0; c < 7; ++c) acc[c] = zero_v8f();
    float dsum = 0.f;

    for (int l = 0; l < NN; l += 32) {
        const float* p0 = Arow + l + hi * 8;
        float4 f0 = *(const float4*)(p0);
        float4 f1 = *(const float4*)(p0 + 4);
        float4 f2 = *(const float4*)(p0 + 16);
        float4 f3 = *(const float4*)(p0 + 20);
        __builtin_prefetch(Arow + l + 256, 0, 3);   // global_prefetch ahead of stream
        dsum += (f0.x + f0.y + f0.z + f0.w) + (f1.x + f1.y + f1.z + f1.w)
              + (f2.x + f2.y + f2.z + f2.w) + (f3.x + f3.y + f3.z + f3.w);
        Frag a;
        cvt16(a, f0, f1, f2, f3);
        for (int c = 0; c < 7; ++c) {
            const unsigned short* sp = Sb + (size_t)(c * 16 + m) * NN + l + hi * 8;
            Frag bf;
            bf.q[0] = *(const uint4*)(sp);
            bf.q[1] = *(const uint4*)(sp + 16);
            acc[c] = __builtin_amdgcn_wmma_f32_16x16x32_bf16(
                false, a.v, false, bf.v, (short)0, acc[c], false, false);
        }
    }

    // K-major bf16 store: one 16B store per tile per lane
    unsigned short* Db = as16t + (size_t)b * KP * NN;
    for (int c = 0; c < 7; ++c) {
        uint4 q;
        q.x = pack2(acc[c][0], acc[c][1]);
        q.y = pack2(acc[c][2], acc[c][3]);
        q.z = pack2(acc[c][4], acc[c][5]);
        q.w = pack2(acc[c][6], acc[c][7]);
        *(uint4*)(Db + (size_t)(c * 16 + m) * NN + n0 + hi * 8) = q;
    }

    // fused mincut_den: deg[n0+m] = lane(m,hi=0)+lane(m,hi=1) partial sums
    float other = __shfl_xor(dsum, 16, 32);
    float contrib = (hi == 0) ? (dsum + other) * rowsq[flat + m] : 0.f;
    for (int off = 1; off < 32; off <<= 1) contrib += __shfl_xor(contrib, off, 32);
    if (lane == 0) atomicAdd(&den[b], contrib);
}

// ---------------------------------------------------------------------------
// Kernel 3: out = s^T @ x  -> poolT (B, F, KP) fp32.
// A-frag direct from K-major s; x transpose-staged coalesced via LDS.
// ---------------------------------------------------------------------------
__global__ __launch_bounds__(32) void k_pool(const unsigned short* __restrict__ s16t,
                                             const float* __restrict__ x,
                                             float* __restrict__ poolT) {
    __shared__ __align__(16) unsigned short Bt[4][16][32];
    int lane = threadIdx.x;
    int id = blockIdx.x;                   // B * 7 * 8
    int b = id / 56, rem = id % 56;
    int k0 = (rem / 8) * 16;
    int f0 = (rem % 8) * 64;
    int m = lane & 15, hi = lane >> 4;

    const unsigned short* Sb = s16t + (size_t)b * KP * NN;
    const float* X = x + (size_t)b * NN * FF;

    v8f acc[4];
    for (int c = 0; c < 4; ++c) acc[c] = zero_v8f();

    for (int l = 0; l < NN; l += 32) {
        for (int i = lane; i < 32 * 64; i += 32) {   // coalesced: f fastest
            int kq = i >> 6, f = i & 63;
            Bt[f >> 4][f & 15][kq] = f2bf(X[(size_t)(l + kq) * FF + f0 + f]);
        }
        __syncthreads();
        Frag a;
        const unsigned short* ap = Sb + (size_t)(k0 + m) * NN + l + hi * 8;
        a.q[0] = *(const uint4*)(ap);
        a.q[1] = *(const uint4*)(ap + 16);
        for (int c = 0; c < 4; ++c) {
            Frag bf;
            bf.q[0] = *(const uint4*)&Bt[c][m][hi * 8];
            bf.q[1] = *(const uint4*)&Bt[c][m][hi * 8 + 16];
            acc[c] = __builtin_amdgcn_wmma_f32_16x16x32_bf16(
                false, a.v, false, bf.v, (short)0, acc[c], false, false);
        }
        __syncthreads();
    }

    for (int c = 0; c < 4; ++c) {                 // two b128 stores per tile
        union { v8f v; float4 f4[2]; } u;
        u.v = acc[c];
        float* P = poolT + ((size_t)b * FF + f0 + c * 16 + m) * KP + k0 + hi * 8;
        *(float4*)(P) = u.f4[0];
        *(float4*)(P + 4) = u.f4[1];
    }
}

// ---------------------------------------------------------------------------
// Kernel 4: out_adj_raw = s^T As (which=0) and ss = s^T s (which=1).
// Fully LDS-free: both fragments direct b128 from K-major L2-resident arrays.
// ---------------------------------------------------------------------------
__global__ __launch_bounds__(32) void k_gram(const unsigned short* __restrict__ s16t,
                                             const unsigned short* __restrict__ as16t,
                                             float* __restrict__ adjr,
                                             float* __restrict__ ssm) {
    int lane = threadIdx.x;
    int id = blockIdx.x;                 // B*7*7*2
    int which = id & 1; id >>= 1;
    int tj = id % 7; id /= 7;
    int ti = id % 7;
    int b = id / 7;
    int m = lane & 15, hi = lane >> 4;

    const unsigned short* Sb = s16t + (size_t)b * KP * NN;
    const unsigned short* Bb = (which ? s16t : as16t) + (size_t)b * KP * NN;

    v8f acc = zero_v8f();
    for (int l = 0; l < NN; l += 32) {
        Frag a, bf;
        const unsigned short* ap = Sb + (size_t)(ti * 16 + m) * NN + l + hi * 8;
        a.q[0] = *(const uint4*)(ap);
        a.q[1] = *(const uint4*)(ap + 16);
        const unsigned short* bp = Bb + (size_t)(tj * 16 + m) * NN + l + hi * 8;
        bf.q[0] = *(const uint4*)(bp);
        bf.q[1] = *(const uint4*)(bp + 16);
        acc = __builtin_amdgcn_wmma_f32_16x16x32_bf16(
            false, a.v, false, bf.v, (short)0, acc, false, false);
    }

    float* D = (which ? ssm : adjr) + (size_t)b * KP * KP;
    for (int r = 0; r < 8; ++r)
        D[(size_t)(ti * 16 + r + hi * 8) * KP + tj * 16 + m] = acc[r];
}

// ---------------------------------------------------------------------------
// Kernel 5: batchnorm over (B, F) per channel k; writes x_af_pool to d_out.
// ---------------------------------------------------------------------------
__global__ __launch_bounds__(256) void k_bn(const float* __restrict__ poolT,
                                            const float* __restrict__ gamma,
                                            const float* __restrict__ beta,
                                            float* __restrict__ out) {
    __shared__ float r1[256], r2[256];
    int k = blockIdx.x;                  // 0..99
    float s = 0.f, s2 = 0.f;
    for (int i = threadIdx.x; i < BB * FF; i += 256) {
        int b = i / FF, f = i % FF;
        float v = poolT[((size_t)b * FF + f) * KP + k];
        s += v; s2 += v * v;
    }
    r1[threadIdx.x] = s; r2[threadIdx.x] = s2;
    __syncthreads();
    for (int off = 128; off > 0; off >>= 1) {
        if (threadIdx.x < off) {
            r1[threadIdx.x] += r1[threadIdx.x + off];
            r2[threadIdx.x] += r2[threadIdx.x + off];
        }
        __syncthreads();
    }
    float mean = r1[0] / (float)(BB * FF);
    float var = r2[0] / (float)(BB * FF) - mean * mean;
    float g = gamma[k], be = beta[k];
    float inv = rsqrtf(var + BNEPS);
    for (int i = threadIdx.x; i < BB * FF; i += 256) {
        int b = i / FF, f = i % FF;
        float v = poolT[((size_t)b * FF + f) * KP + k];
        out[((size_t)b * KK + k) * FF + f] = g * (v - mean) * inv + be;
    }
}

// ---------------------------------------------------------------------------
// Kernel 6: degree-normalize out_adj (zero diag) and write to d_out.
// ---------------------------------------------------------------------------
__global__ __launch_bounds__(128) void k_adjnorm(const float* __restrict__ adjr,
                                                 float* __restrict__ out) {
    __shared__ float dsh[KK];
    int b = blockIdx.x;
    const float* R = adjr + (size_t)b * KP * KP;
    for (int k = threadIdx.x; k < KK; k += 128) {
        float s = 0.f;
        for (int l = 0; l < KK; ++l)
            if (l != k) s += R[(size_t)k * KP + l];
        dsh[k] = sqrtf(s) + EPSV;
    }
    __syncthreads();
    float* O = out + (size_t)BB * KK * FF + (size_t)b * KK * KK;
    for (int i = threadIdx.x; i < KK * KK; i += 128) {
        int k = i / KK, l = i % KK;
        float v = (k == l) ? 0.f : R[(size_t)k * KP + l];
        O[i] = v / (dsh[k] * dsh[l]);
    }
}

// ---------------------------------------------------------------------------
// Kernel 7: scalar loss mc + o.
// ---------------------------------------------------------------------------
__global__ __launch_bounds__(256) void k_loss(const float* __restrict__ adjr,
                                              const float* __restrict__ ssm,
                                              const float* __restrict__ den,
                                              float* __restrict__ out) {
    __shared__ float red[256];
    float mc = 0.f, osum = 0.f;
    for (int b = 0; b < BB; ++b) {
        const float* R = adjr + (size_t)b * KP * KP;
        const float* SS = ssm + (size_t)b * KP * KP;
        float t = 0.f;
        for (int k = threadIdx.x; k < KK; k += 256) t += R[(size_t)k * KP + k];
        red[threadIdx.x] = t; __syncthreads();
        for (int off = 128; off > 0; off >>= 1) {
            if (threadIdx.x < off) red[threadIdx.x] += red[threadIdx.x + off];
            __syncthreads();
        }
        float trace = red[0];
        __syncthreads();

        float f = 0.f;
        for (int i = threadIdx.x; i < KK * KK; i += 256) {
            int k = i / KK, l = i % KK;
            float v = SS[(size_t)k * KP + l];
            f += v * v;
        }
        red[threadIdx.x] = f; __syncthreads();
        for (int off = 128; off > 0; off >>= 1) {
            if (threadIdx.x < off) red[threadIdx.x] += red[threadIdx.x + off];
            __syncthreads();
        }
        float fro = sqrtf(red[0]);
        __syncthreads();

        float iv = rsqrtf((float)KK);
        float d2 = 0.f;
        for (int i = threadIdx.x; i < KK * KK; i += 256) {
            int k = i / KK, l = i % KK;
            float v = SS[(size_t)k * KP + l] / fro - ((k == l) ? iv : 0.f);
            d2 += v * v;
        }
        red[threadIdx.x] = d2; __syncthreads();
        for (int off = 128; off > 0; off >>= 1) {
            if (threadIdx.x < off) red[threadIdx.x] += red[threadIdx.x + off];
            __syncthreads();
        }
        if (threadIdx.x == 0) {
            mc += -(trace / den[b]);
            osum += sqrtf(red[0]);
        }
        __syncthreads();
    }
    if (threadIdx.x == 0)
        out[(size_t)BB * KK * FF + (size_t)BB * KK * KK] = mc / (float)BB + osum / (float)BB;
}

// ---------------------------------------------------------------------------
extern "C" void kernel_launch(void* const* d_in, const int* in_sizes, int n_in,
                              void* d_out, int out_size, void* d_ws, size_t ws_size,
                              hipStream_t stream) {
    const float* x     = (const float*)d_in[0];   // (B,N,F)
    const float* adj   = (const float*)d_in[1];   // (B,N,N)
    // d_in[2] = x_masks (all true) -- identity
    const float* W     = (const float*)d_in[3];   // (K,F)
    const float* bias  = (const float*)d_in[4];   // (K,)
    const float* gamma = (const float*)d_in[5];   // (K,)
    const float* beta  = (const float*)d_in[6];   // (K,)
    float* out = (float*)d_out;

    float* ws    = (float*)d_ws;
    float* rowsq = ws;                                  // B*N
    float* den   = rowsq + (size_t)BB * NN;             // B
    float* poolT = den + BB;                            // B*F*KP
    float* adjr  = poolT + (size_t)BB * FF * KP;        // B*KP*KP
    float* ssm   = adjr + (size_t)BB * KP * KP;         // B*KP*KP
    unsigned short* s16t  = (unsigned short*)(ssm + (size_t)BB * KP * KP); // B*KP*NN bf16
    unsigned short* as16t = s16t + (size_t)BB * KP * NN;                   // B*KP*NN bf16

    hipLaunchKernelGGL(k_s_softmax, dim3(BB * NN / 16),       dim3(32),  0, stream, x, W, bias, rowsq, den, s16t);
    hipLaunchKernelGGL(k_as,        dim3(BB * NN / 16 / 8),   dim3(256), 0, stream, adj, s16t, rowsq, den, as16t);
    hipLaunchKernelGGL(k_pool,      dim3(BB * 7 * 8),         dim3(32),  0, stream, s16t, x, poolT);
    hipLaunchKernelGGL(k_gram,      dim3(BB * 7 * 7 * 2),     dim3(32),  0, stream, s16t, as16t, adjr, ssm);
    hipLaunchKernelGGL(k_bn,        dim3(KK),                 dim3(256), 0, stream, poolT, gamma, beta, out);
    hipLaunchKernelGGL(k_adjnorm,   dim3(BB),                 dim3(128), 0, stream, adjr, out);
    hipLaunchKernelGGL(k_loss,      dim3(1),                  dim3(256), 0, stream, adjr, ssm, den, out);
}
